// RetNet_39719857553619
// MI455X (gfx1250) — matrix operand order repaired and checked by reading, not compiled, not run on hardware
//
#include <hip/hip_runtime.h>
#include <hip/hip_bf16.h>
#include <math.h>

#define N_LAYERS 4
#define HID      512
#define FFND     1024
#define NHEADS   4
#define HDIM     128
#define NB       2
#define NS       2048
#define NBS      (NB * NS)   // 4096 rows

typedef __attribute__((ext_vector_type(16))) __bf16       v16bf;
typedef __attribute__((ext_vector_type(8)))  float        v8f;
typedef __attribute__((ext_vector_type(8)))  unsigned int v8u;

// pack two fp32 into a bf16x2 dword (lowers to v_cvt_pk_bf16_f32)
static __device__ __forceinline__ unsigned pkbf(float a, float b)
{
    unsigned short lo = __builtin_bit_cast(unsigned short, (__bf16)a);
    unsigned short hi = __builtin_bit_cast(unsigned short, (__bf16)b);
    return (unsigned)lo | ((unsigned)hi << 16);
}

// read one lane's fragment: 32 contiguous bytes of pre-swizzled LDS
static __device__ __forceinline__ v16bf ldfrag(const unsigned* p)
{
    return __builtin_bit_cast(v16bf, *(const v8u*)p);
}

// A-matrix pair (k,k+1), k even -> (hi, slot v) per ISA 7.12.2 16-bit A layout
#define A_HI(k) (((k) >> 3) & 1)
#define A_V(k)  ((((k) & 16) >> 2) + (((k) & 7) >> 1))
// B-matrix pair (k,k+1), k even -> (hi, slot v)
#define B_HI(k) (((k) >> 3) & 1)
#define B_V(k)  ((((k) & 16) >> 2) + (((k) & 7) >> 1))

// ---------------------------------------------------------------------------
// WMMA GEMM, 64x64 macro-tile per 4-wave workgroup.
// Tiles staged in LDS as bf16, pre-swizzled into WMMA fragment order.
// C[M x N] = act(A[M x K] * B[K x N] + bias) + res ; K % 32 == 0, M,N % 64 == 0
// ---------------------------------------------------------------------------
__global__ __launch_bounds__(128)
void gemm64_wmma(const float* __restrict__ A, int lda,
                 const float* __restrict__ Bm, int ldb,
                 const float* __restrict__ bias,
                 const float* __restrict__ res,
                 float* __restrict__ C, int ldc,
                 int K, int act)
{
    // [subtile/nb][lane][slot] packed bf16x2
    __shared__ __attribute__((aligned(32))) unsigned Asw[4 * 32 * 8];
    __shared__ __attribute__((aligned(32))) unsigned Bsw[4 * 32 * 8];

    const int tid  = threadIdx.x;
    const int wave = tid >> 5;
    const int lane = tid & 31;
    const int l16  = lane & 15;
    const int hi   = lane >> 4;

    const size_t m0 = (size_t)blockIdx.x * 64;
    const size_t n0 = (size_t)blockIdx.y * 64;

    v8f acc[4];
#pragma unroll
    for (int i = 0; i < 4; ++i) { v8f z = {}; acc[i] = z; }

    // B loader mapping: 16 k-pair rows x 8 threads covering 8 cols each
    const int k2 = tid >> 3;            // 0..15 -> k = 2*k2
    const int n8 = (tid & 7) * 8;       // first of 8 columns

    for (int k0 = 0; k0 < K; k0 += 32) {
        // ---- stage A (64x32) as swizzled bf16 fragments ----
#pragma unroll
        for (int i = 0; i < 4; ++i) {
            int flat = i * 512 + tid * 4;                 // element index in tile
            int ar = flat >> 5, ac = flat & 31;
            float4 t = *(const float4*)(A + (m0 + ar) * lda + k0 + ac);
            int sub = ar >> 4, m = ar & 15;
            Asw[((sub * 32) + m + 16 * A_HI(ac)) * 8 + A_V(ac)]         = pkbf(t.x, t.y);
            Asw[((sub * 32) + m + 16 * A_HI(ac + 2)) * 8 + A_V(ac + 2)] = pkbf(t.z, t.w);
        }
        // ---- stage B (32x64) as swizzled bf16 fragments ----
        {
            const float* r0 = Bm + (size_t)(k0 + 2 * k2) * ldb + n0 + n8;
            float4 b0 = *(const float4*)(r0);
            float4 b1 = *(const float4*)(r0 + 4);
            float4 c0 = *(const float4*)(r0 + ldb);
            float4 c1 = *(const float4*)(r0 + ldb + 4);
            const int bhi = B_HI(2 * k2), bv = B_V(2 * k2);
            float t0[8] = { b0.x, b0.y, b0.z, b0.w, b1.x, b1.y, b1.z, b1.w };
            float t1[8] = { c0.x, c0.y, c0.z, c0.w, c1.x, c1.y, c1.z, c1.w };
#pragma unroll
            for (int j = 0; j < 8; ++j) {
                int n = n8 + j;
                Bsw[(((n >> 4) * 32) + (n & 15) + 16 * bhi) * 8 + bv] = pkbf(t0[j], t1[j]);
            }
        }
        if (k0 + 32 < K) {
            int flat = tid * 4;
            __builtin_prefetch(A + (m0 + (flat >> 5)) * lda + k0 + 32 + (flat & 31), 0, 0);
            __builtin_prefetch(Bm + (size_t)(k0 + 32 + 2 * k2) * ldb + n0 + n8, 0, 0);
        }
        __syncthreads();

        // ---- compute: 1 A fragment reused across 4 B fragments ----
        v16bf a = ldfrag(&Asw[(wave * 32 + lane) * 8]);
#pragma unroll
        for (int nb = 0; nb < 4; ++nb) {
            v16bf b = ldfrag(&Bsw[(nb * 32 + lane) * 8]);
            acc[nb] = __builtin_amdgcn_wmma_f32_16x16x32_bf16(false, a, false, b,
                                                              (short)0, acc[nb], false, false);
        }
        __syncthreads();
    }

    // ---- epilogue ----
#pragma unroll
    for (int nb = 0; nb < 4; ++nb) {
        int n = (int)n0 + nb * 16 + l16;
#pragma unroll
        for (int r = 0; r < 8; ++r) {
            size_t m = m0 + wave * 16 + r + (hi ? 8 : 0);
            float v = acc[nb][r];
            if (bias) v += bias[n];
            if (act == 1) v = 0.5f * v * (1.0f + erff(v * 0.70710678118654752f));
            if (res) v += res[m * ldc + n];
            C[m * ldc + n] = v;
        }
    }
}

// ---------------------------------------------------------------------------
// LayerNorm over HID=512, one wave per row
// ---------------------------------------------------------------------------
__global__ __launch_bounds__(32)
void ln_kernel(const float* __restrict__ X, const float* __restrict__ w,
               const float* __restrict__ b, float* __restrict__ out)
{
    const size_t row = blockIdx.x;
    const int lane = threadIdx.x;
    const float* x = X + row * HID;
    float vals[16];
    float s = 0.f, s2 = 0.f;
#pragma unroll
    for (int i = 0; i < 16; ++i) {
        float v = x[lane * 16 + i];
        vals[i] = v; s += v; s2 += v * v;
    }
#pragma unroll
    for (int off = 16; off; off >>= 1) {
        s  += __shfl_xor(s,  off, 32);
        s2 += __shfl_xor(s2, off, 32);
    }
    float mu  = s * (1.0f / HID);
    float var = s2 * (1.0f / HID) - mu * mu;
    float inv = rsqrtf(var + 1e-5f);
#pragma unroll
    for (int i = 0; i < 16; ++i) {
        int e = lane * 16 + i;
        out[row * HID + e] = (vals[i] - mu) * inv * w[e] + b[e];
    }
}

// ---------------------------------------------------------------------------
// XPos rotary on Q/K buffers laid out as [h][b*S][HDIM]; one thread per pair
// ---------------------------------------------------------------------------
__global__ void xpos_kernel(float* __restrict__ Qb, int down)
{
    size_t idx  = (size_t)blockIdx.x * blockDim.x + threadIdx.x;
    size_t rowm = idx >> 6;
    int    j    = (int)(idx & 63);
    int    s    = (int)(rowm % NS);

    float pos   = (float)s;
    float hd    = (float)HDIM;
    float sj    = (2.0f * j + 0.4f * hd) / (1.4f * hd);
    float scale = exp2f((pos / 512.0f) * log2f(sj));
    if (down) scale = 1.0f / scale;
    float inv_freq = powf(10000.0f, -(float)j / 64.0f);
    float ang = pos * inv_freq;
    float sn = sinf(ang), cs = cosf(ang);

    float* p = Qb + rowm * HDIM + 2 * j;
    float x1 = p[0], x2 = p[1];
    p[0] = x1 * cs * scale - x2 * sn * scale;
    p[1] = x2 * cs * scale + x1 * sn * scale;
}

// ---------------------------------------------------------------------------
// Retention: 4-wave blocks; 64 query rows share each 32x128 K/V chunk, staged
// in LDS as pre-swizzled bf16 WMMA fragments.
// ---------------------------------------------------------------------------
__global__ __launch_bounds__(128)
void retention_kernel(const float* __restrict__ Qb, const float* __restrict__ Kb,
                      const float* __restrict__ Vb, float* __restrict__ Yh)
{
    // Ksw[g*4+kk][lane][slot] : B-fragments of K^T  (2 key groups x 4 e-slices)
    // Vsw[nt][lane][slot]     : B-fragments of V    (8 e-tiles)
    __shared__ __attribute__((aligned(32))) unsigned Ksw[8 * 32 * 8];
    __shared__ __attribute__((aligned(32))) unsigned Vsw[8 * 32 * 8];
    __shared__ float Ss[4][16 * 32];

    const int tid  = threadIdx.x;
    const int wave = tid >> 5;
    const int lane = tid & 31;
    const int l16  = lane & 15;
    const int hi   = lane >> 4;

    const int qt = blockIdx.x * 4 + wave;
    const int bh = blockIdx.y;
    const int b = bh / NHEADS, h = bh % NHEADS;
    const size_t base = ((size_t)h * NBS + (size_t)b * NS) * HDIM;
    const float* Q = Qb + base;
    const float* K = Kb + base;
    const float* V = Vb + base;

    const float lg0 = logf(1.0f / 32.0f), lg1 = logf(1.0f / 512.0f);
    const float gm = 1.0f - expf(lg0 + (lg1 - lg0) * (float)h / (float)(NHEADS - 1));
    const float log2g = log2f(gm);

    // Q tile as 4 A-fragments (K-dim = 128 in slices of 32), from global
    v16bf aq[4];
    const size_t qrow = (size_t)qt * 16 + l16;
#pragma unroll
    for (int kk = 0; kk < 4; ++kk)
#pragma unroll
        for (int v = 0; v < 8; ++v) {
            int e = kk * 32 + ((v < 4) ? 2 * v : 16 + 2 * (v - 4)) + (hi ? 8 : 0);
            float2 t = *(const float2*)(Q + qrow * HDIM + e);
            aq[kk][2 * v]     = (__bf16)t.x;
            aq[kk][2 * v + 1] = (__bf16)t.y;
        }

    v8f ret[8];
#pragma unroll
    for (int i = 0; i < 8; ++i) { v8f z = {}; ret[i] = z; }

    const int nch  = 2 * blockIdx.x + 2;
    const int cmax = (qt * 16 + 15) >> 5;
    float* sS = Ss[wave];

    // V loader mapping: 16 t-pair rows x 8 threads x 16 cols
    const int t2  = tid >> 3;           // key pair row: t = 2*t2
    const int e16 = (tid & 7) * 16;     // first of 16 e-columns

    for (int c = 0; c < nch; ++c) {
        const int t0 = c * 32;

        // ---- stage K chunk: pairs along e are thread-local ----
#pragma unroll
        for (int i = 0; i < 8; ++i) {
            int flat = i * 512 + tid * 4;                  // 32x128 tile
            int t = flat >> 7, e = flat & 127;
            float4 kv = *(const float4*)(K + (size_t)(t0 + t) * HDIM + e);
            int g = t >> 4, n = t & 15;
            int kk = e >> 5, er = e & 31;
            Ksw[(((g * 4 + kk) * 32) + n + 16 * B_HI(er)) * 8 + B_V(er)]         = pkbf(kv.x, kv.y);
            Ksw[(((g * 4 + kk) * 32) + n + 16 * B_HI(er + 2)) * 8 + B_V(er + 2)] = pkbf(kv.z, kv.w);
        }
        // ---- stage V chunk: pairs along t need two rows per thread ----
        {
            const float* r0 = V + (size_t)(t0 + 2 * t2) * HDIM + e16;
            const int vhi = B_HI(2 * t2), vv = B_V(2 * t2);
#pragma unroll
            for (int q4 = 0; q4 < 4; ++q4) {
                float4 a0 = *(const float4*)(r0 + q4 * 4);
                float4 a1 = *(const float4*)(r0 + HDIM + q4 * 4);
                float u0[4] = { a0.x, a0.y, a0.z, a0.w };
                float u1[4] = { a1.x, a1.y, a1.z, a1.w };
#pragma unroll
                for (int j = 0; j < 4; ++j) {
                    int e = e16 + q4 * 4 + j;
                    Vsw[(((e >> 4) * 32) + (e & 15) + 16 * vhi) * 8 + vv] = pkbf(u0[j], u1[j]);
                }
            }
        }
        if (c + 1 < nch) {
            int flat = tid * 4;
            __builtin_prefetch(K + (size_t)(t0 + 32 + (flat >> 7)) * HDIM + (flat & 127), 0, 0);
            __builtin_prefetch(V + (size_t)(t0 + 32 + 2 * t2) * HDIM + e16, 0, 0);
        }
        __syncthreads();

        if (c <= cmax) {
            // ---- scores: two 16x16 f32 tiles over keys [t0, t0+32) ----
#pragma unroll
            for (int g = 0; g < 2; ++g) {
                v8f sacc = {};
#pragma unroll
                for (int kk = 0; kk < 4; ++kk) {
                    v16bf bk = ldfrag(&Ksw[((g * 4 + kk) * 32 + lane) * 8]);
                    sacc = __builtin_amdgcn_wmma_f32_16x16x32_bf16(false, aq[kk], false, bk,
                                                                   (short)0, sacc, false, false);
                }
#pragma unroll
                for (int r = 0; r < 8; ++r) {
                    int m = r + hi * 8;
                    int s = qt * 16 + m;
                    int t = t0 + g * 16 + l16;
                    int d = s - t;
                    float w = (d >= 0) ? exp2f((float)d * log2g) : 0.0f;
                    sS[m * 32 + g * 16 + l16] = sacc[r] * w;
                }
            }

            // ---- reload scores as a 16x32 A-fragment ----
            v16bf as;
#pragma unroll
            for (int v = 0; v < 8; ++v) {
                int k = ((v < 4) ? 2 * v : 16 + 2 * (v - 4)) + (hi ? 8 : 0);
                float2 t = *(const float2*)&sS[l16 * 32 + k];
                as[2 * v]     = (__bf16)t.x;
                as[2 * v + 1] = (__bf16)t.y;
            }

            // ---- ret[16x128] += scores[16x32] x V[32x128] ----
#pragma unroll
            for (int nt = 0; nt < 8; ++nt) {
                v16bf bv = ldfrag(&Vsw[(nt * 32 + lane) * 8]);
                ret[nt] = __builtin_amdgcn_wmma_f32_16x16x32_bf16(false, as, false, bv,
                                                                  (short)0, ret[nt], false, false);
            }
        }
        __syncthreads();
    }

    // ---- write back merged-head layout Yh[b*S + s][h*128 + e] ----
#pragma unroll
    for (int nt = 0; nt < 8; ++nt)
#pragma unroll
        for (int r = 0; r < 8; ++r) {
            int m = r + hi * 8;
            size_t row = (size_t)b * NS + (size_t)qt * 16 + m;
            Yh[row * HID + h * HDIM + nt * 16 + l16] = ret[nt][r];
        }
}

// ---------------------------------------------------------------------------
// Group norm over heads + SiLU gating:  G <- silu(G) * (GN(Yh)*w + b)
// ---------------------------------------------------------------------------
__global__ __launch_bounds__(32)
void msr_kernel(float* __restrict__ G, const float* __restrict__ Yh,
                const float* __restrict__ gnw, const float* __restrict__ gnb)
{
    const size_t row = blockIdx.x;
    const int lane = threadIdx.x;
    const float* y = Yh + row * HID;
    float vals[16];
    float s = 0.f, s2 = 0.f;
#pragma unroll
    for (int i = 0; i < 16; ++i) {
        float v = y[lane * 16 + i];
        vals[i] = v; s += v; s2 += v * v;
    }
#pragma unroll
    for (int off = 4; off; off >>= 1) {
        s  += __shfl_xor(s,  off, 32);
        s2 += __shfl_xor(s2, off, 32);
    }
    float mu  = s * (1.0f / HDIM);
    float var = s2 * (1.0f / HDIM) - mu * mu;
    float inv = rsqrtf(var + 1e-5f);
#pragma unroll
    for (int i = 0; i < 16; ++i) {
        int e = lane * 16 + i;
        float yn = (vals[i] - mu) * inv * gnw[e] + gnb[e];
        float g  = G[row * HID + e];
        float sig = 1.0f / (1.0f + expf(-g));
        G[row * HID + e] = g * sig * yn;
    }
}

// ---------------------------------------------------------------------------
extern "C" void kernel_launch(void* const* d_in, const int* in_sizes, int n_in,
                              void* d_out, int out_size, void* d_ws, size_t ws_size,
                              hipStream_t stream)
{
    const float* X0   = (const float*)d_in[0];
    const float* Wq   = (const float*)d_in[1];
    const float* Wk   = (const float*)d_in[2];
    const float* Wv   = (const float*)d_in[3];
    const float* Wg   = (const float*)d_in[4];
    const float* Wo   = (const float*)d_in[5];
    const float* gn_w = (const float*)d_in[6];
    const float* gn_b = (const float*)d_in[7];
    const float* l1w  = (const float*)d_in[8];
    const float* l1b  = (const float*)d_in[9];
    const float* l2w  = (const float*)d_in[10];
    const float* l2b  = (const float*)d_in[11];
    const float* f1w  = (const float*)d_in[12];
    const float* f1b  = (const float*)d_in[13];
    const float* f2w  = (const float*)d_in[14];
    const float* f2b  = (const float*)d_in[15];

    float* ws = (float*)d_ws;
    const size_t R = (size_t)NBS * HID;
    float* Xb = ws;
    float* Xn = Xb + R;
    float* Qb = Xn + R;
    float* Kb = Qb + R;
    float* Vb = Kb + R;
    float* Gb = Vb + R;
    float* Yh = Gb + R;
    float* Yb = Yh + R;
    float* H1 = Yb + R;                   // NBS * FFND

    hipMemcpyAsync(Xb, X0, R * sizeof(float), hipMemcpyDeviceToDevice, stream);

    const dim3 blk1(32), blk4(128);
    for (int l = 0; l < N_LAYERS; ++l) {
        ln_kernel<<<NBS, blk1, 0, stream>>>(Xb, l1w + l * HID, l1b + l * HID, Xn);

        for (int h = 0; h < NHEADS; ++h) {
            const size_t woff = ((size_t)l * NHEADS + h) * HID * HDIM;
            gemm64_wmma<<<dim3(NBS / 64, HDIM / 64), blk4, 0, stream>>>(
                Xn, HID, Wq + woff, HDIM, nullptr, nullptr,
                Qb + (size_t)h * NBS * HDIM, HDIM, HID, 0);
            gemm64_wmma<<<dim3(NBS / 64, HDIM / 64), blk4, 0, stream>>>(
                Xn, HID, Wk + woff, HDIM, nullptr, nullptr,
                Kb + (size_t)h * NBS * HDIM, HDIM, HID, 0);
            gemm64_wmma<<<dim3(NBS / 64, HDIM / 64), blk4, 0, stream>>>(
                Xn, HID, Wv + woff, HDIM, nullptr, nullptr,
                Vb + (size_t)h * NBS * HDIM, HDIM, HID, 0);
        }

        gemm64_wmma<<<dim3(NBS / 64, HID / 64), blk4, 0, stream>>>(
            Xn, HID, Wg + (size_t)l * HID * HID, HID, nullptr, nullptr,
            Gb, HID, HID, 0);

        const int npairs = NB * NHEADS * NS * (HDIM / 2);
        xpos_kernel<<<npairs / 256, 256, 0, stream>>>(Qb, 0);
        xpos_kernel<<<npairs / 256, 256, 0, stream>>>(Kb, 1);

        retention_kernel<<<dim3(NS / 64, NB * NHEADS), blk4, 0, stream>>>(Qb, Kb, Vb, Yh);

        msr_kernel<<<NBS, blk1, 0, stream>>>(Gb, Yh, gn_w + l * HID, gn_b + l * HID);

        gemm64_wmma<<<dim3(NBS / 64, HID / 64), blk4, 0, stream>>>(
            Gb, HID, Wo + (size_t)l * HID * HID, HID, nullptr, Xb,
            Yb, HID, HID, 0);

        ln_kernel<<<NBS, blk1, 0, stream>>>(Yb, l2w + l * HID, l2b + l * HID, Xn);

        gemm64_wmma<<<dim3(NBS / 64, FFND / 64), blk4, 0, stream>>>(
            Xn, HID, f1w + (size_t)l * HID * FFND, FFND, f1b + (size_t)l * FFND,
            nullptr, H1, FFND, HID, 1);

        gemm64_wmma<<<dim3(NBS / 64, HID / 64), blk4, 0, stream>>>(
            H1, FFND, f2w + (size_t)l * FFND * HID, HID, f2b + (size_t)l * HID,
            Yb, Xb, HID, FFND, 0);
    }

    hipMemcpyAsync(d_out, Xb, R * sizeof(float), hipMemcpyDeviceToDevice, stream);
}
